// GenFlow_unit_mask_78426102825582
// MI455X (gfx1250) — compile-verified, matrix operand order checked by reading, not run on hardware
//
#include <hip/hip_runtime.h>
#include <hip/hip_bf16.h>
#include <stdint.h>
#include <stddef.h>

typedef __attribute__((ext_vector_type(16))) _Float16 v16h;
typedef __attribute__((ext_vector_type(8)))  float    v8f;

static inline unsigned cdivh(size_t a, size_t b) { return (unsigned)((a + b - 1) / b); }

// ---------------------------------------------------------------------------
// WMMA GEMM: Y = relu(X[R,K] @ W + bias[N]).
// X row-major f16 (A-fragment loads coalesce to 2x b128 per lane per k-tile).
// Wp pre-packed in B-fragment order: [((kt*NT + nt)*32 + lane)*16 + e], so each
// lane's fragment is one contiguous 32B load. One wave computes a 16x32 tile
// (two 16x16 WMMA tiles sharing the A fragment). K templated -> full unroll,
// back-to-back v_wmma_f32_16x16x32_f16. Write mode templated -> branchless
// epilogue.
// ---------------------------------------------------------------------------
template <int KT, bool WH, bool WF>
__global__ void k_wmma_gemm2(const _Float16* __restrict__ X,
                             const _Float16* __restrict__ Wp,
                             const float* __restrict__ bias,
                             _Float16* __restrict__ Yh,
                             float* __restrict__ Yf,
                             int R, int N) {
  constexpr int K = KT * 32;
  const int lane = threadIdx.x & 31;
  const int row0 = blockIdx.x << 4;
  const int col0 = blockIdx.y << 5;   // two 16-col tiles per wave
  const int l15  = lane & 15;
  const int hi   = lane >> 4;         // K-offset split per ISA 7.12.2 A layout
  const int NT   = N >> 4;
  const int nt0  = col0 >> 4;
  const _Float16* xr = X + (size_t)(row0 + l15) * K + hi * 8;
  v8f acc0 = {}, acc1 = {};
#pragma unroll
  for (int kt = 0; kt < KT; ++kt) {
    v16h a;
    const _Float16* xk = xr + kt * 32;
#pragma unroll
    for (int e = 0; e < 8; ++e) a[e] = xk[e];          // K = hi*8 + 0..7
#pragma unroll
    for (int e = 0; e < 8; ++e) a[8 + e] = xk[16 + e]; // K = 16 + hi*8 + 0..7
    const v16h b0 = *(const v16h*)(Wp + (((size_t)kt * NT + nt0)     * 32 + lane) * 16);
    const v16h b1 = *(const v16h*)(Wp + (((size_t)kt * NT + nt0 + 1) * 32 + lane) * 16);
    acc0 = __builtin_amdgcn_wmma_f32_16x16x32_f16(false, a, false, b0, (short)0, acc0, false, false);
    acc1 = __builtin_amdgcn_wmma_f32_16x16x32_f16(false, a, false, b1, (short)0, acc1, false, false);
  }
  const float bs0 = bias[col0 + l15];
  const float bs1 = bias[col0 + 16 + l15];
#pragma unroll
  for (int v = 0; v < 8; ++v) {
    // C/D layout: VGPR v -> M = v + 8*hi, N = lane&15
    const int r = row0 + v + hi * 8;
    const float v0 = fmaxf(acc0[v] + bs0, 0.0f);
    const float v1 = fmaxf(acc1[v] + bs1, 0.0f);
    const size_t o0 = (size_t)r * N + col0 + l15;
    if (WH) { Yh[o0] = (_Float16)v0; Yh[o0 + 16] = (_Float16)v1; }
    if (WF) { Yf[o0] = v0;           Yf[o0 + 16] = v1; }
  }
}

// pack weight f32 [Cin,Cout] into B-fragment-ordered f16, zero-padded K rows
__global__ void k_pack_w_frag(const float* __restrict__ W, _Float16* __restrict__ Wp,
                              int Cin, int Cout, int KT) {
  const int NT = Cout >> 4;
  const size_t t = (size_t)blockIdx.x * blockDim.x + threadIdx.x;
  const size_t total = (size_t)KT * NT * 32 * 16;
  if (t >= total) return;
  const int e    = (int)(t & 15);
  const int lane = (int)((t >> 4) & 31);
  const size_t blk = t >> 9;
  const int nt = (int)(blk % NT);
  const int kt = (int)(blk / NT);
  const int hi = lane >> 4;
  const int v  = e >> 1;
  const int kb = (v < 4) ? (hi * 8 + 2 * v) : (16 + hi * 8 + 2 * (v - 4));
  const int k  = kt * 32 + kb + (e & 1);
  const int n  = nt * 16 + (lane & 15);
  Wp[t] = (_Float16)(k < Cin ? W[(size_t)k * Cout + n] : 0.0f);
}

// ---------------------------------------------------------------------------
// Farthest point sampling: one block per batch. dist[] kept in LDS.
// ---------------------------------------------------------------------------
__global__ void k_fps(const float* __restrict__ xyz, int N, int npoint,
                      int* __restrict__ idx) {
  const int b = blockIdx.x;
  const float* p = xyz + (size_t)b * N * 3;
  int* out = idx + (size_t)b * npoint;
  extern __shared__ float smem[];
  float* dist = smem;
  float* redv = smem + N;
  int*   redi = (int*)(redv + blockDim.x);
  __shared__ int s_far;
  __shared__ float s_cx, s_cy, s_cz;
  for (int i = threadIdx.x; i < N; i += blockDim.x) dist[i] = 1e10f;
  if (threadIdx.x == 0) s_far = 0;
  __syncthreads();
  for (int it = 0; it < npoint; ++it) {
    if (threadIdx.x == 0) {
      const int far = s_far;
      out[it] = far;
      s_cx = p[far * 3 + 0]; s_cy = p[far * 3 + 1]; s_cz = p[far * 3 + 2];
    }
    __syncthreads();
    const float cx = s_cx, cy = s_cy, cz = s_cz;
    float bestv = -1.0f; int besti = 0;
    for (int i = threadIdx.x; i < N; i += blockDim.x) {
      const float dx = p[i * 3 + 0] - cx;
      const float dy = p[i * 3 + 1] - cy;
      const float dz = p[i * 3 + 2] - cz;
      float d = dx * dx + dy * dy + dz * dz;
      d = fminf(dist[i], d);
      dist[i] = d;
      if (d > bestv) { bestv = d; besti = i; }
    }
    redv[threadIdx.x] = bestv;
    redi[threadIdx.x] = besti;
    __syncthreads();
    for (int s = blockDim.x >> 1; s > 0; s >>= 1) {
      if ((int)threadIdx.x < s) {
        const float ov = redv[threadIdx.x + s];
        const int   oi = redi[threadIdx.x + s];
        if (ov > redv[threadIdx.x] ||
            (ov == redv[threadIdx.x] && oi < redi[threadIdx.x])) {
          redv[threadIdx.x] = ov; redi[threadIdx.x] = oi;
        }
      }
      __syncthreads();
    }
    if (threadIdx.x == 0) s_far = redi[0];
    __syncthreads();
  }
}

// ---------------------------------------------------------------------------
// Brute-force KNN: one thread per query point, register insertion heap.
// ---------------------------------------------------------------------------
template <int KN>
__global__ void k_knn(const float* __restrict__ q, const float* __restrict__ r,
                      int B, int M, int N, int* __restrict__ nidx) {
  const int t = blockIdx.x * blockDim.x + threadIdx.x;
  if (t >= B * M) return;
  const int b = t / M;
  const float* qp = q + (size_t)t * 3;
  const float* rp = r + (size_t)b * N * 3;
  const float qx = qp[0], qy = qp[1], qz = qp[2];
  float bd[KN]; int bi[KN];
#pragma unroll
  for (int i = 0; i < KN; ++i) { bd[i] = 3.4e38f; bi[i] = 0; }
  for (int n = 0; n < N; ++n) {
    const float dx = rp[n * 3 + 0] - qx;
    const float dy = rp[n * 3 + 1] - qy;
    const float dz = rp[n * 3 + 2] - qz;
    const float d = dx * dx + dy * dy + dz * dz;
    if (d < bd[KN - 1]) {
      bd[KN - 1] = d; bi[KN - 1] = n;
#pragma unroll
      for (int i = KN - 2; i >= 0; --i) {
        if (bd[i + 1] < bd[i]) {
          const float tv = bd[i]; bd[i] = bd[i + 1]; bd[i + 1] = tv;
          const int   ti = bi[i]; bi[i] = bi[i + 1]; bi[i + 1] = ti;
        } else break;
      }
    }
  }
#pragma unroll
  for (int i = 0; i < KN; ++i) nidx[(size_t)t * KN + i] = bi[i];
}

// ---------------------------------------------------------------------------
// Build grouped features: X[b,m,j,:] = [rxyz[nid]-qxyz[m] (3) | rfeat[nid] (Cf) | 0]
// written as f16 row-major [B*M*KN, Kpad].
// ---------------------------------------------------------------------------
__global__ void k_group(const float* __restrict__ rxyz,
                        const float* __restrict__ rfeat, int Cf,
                        const float* __restrict__ qxyz,
                        const int* __restrict__ nidx,
                        int B, int M, int KN, int N,
                        _Float16* __restrict__ X, int Kpad) {
  const size_t t = (size_t)blockIdx.x * blockDim.x + threadIdx.x;
  const size_t total = (size_t)B * M * KN * Kpad;
  if (t >= total) return;
  const int col = (int)(t % Kpad);
  const size_t row = t / Kpad;
  const size_t bm = row / KN;
  const int m = (int)(bm % M);
  const int b = (int)(bm / M);
  const int nid = nidx[row];
  float v = 0.0f;
  if (col < 3) {
    v = rxyz[((size_t)b * N + nid) * 3 + col] - qxyz[((size_t)b * M + m) * 3 + col];
  } else if (col < 3 + Cf) {
    v = rfeat[((size_t)b * N + nid) * Cf + (col - 3)];
  }
  X[t] = (_Float16)v;
}

// max over KN neighbors: H[BM*KN, C] (f16) -> out[BM, C] (f32)
__global__ void k_maxpool(const _Float16* __restrict__ H, float* __restrict__ out,
                          int BM, int KN, int C) {
  const size_t t = (size_t)blockIdx.x * blockDim.x + threadIdx.x;
  if (t >= (size_t)BM * C) return;
  const int c = (int)(t % C);
  const size_t bm = t / C;
  float best = -3.4e38f;
  for (int j = 0; j < KN; ++j) {
    best = fmaxf(best, (float)H[((size_t)bm * KN + j) * C + c]);
  }
  out[t] = best;
}

// concat two f32 feature mats -> f16 padded
__global__ void k_concat2(const float* __restrict__ A, int Ca,
                          const float* __restrict__ Bm, int Cb,
                          _Float16* __restrict__ X, int R, int Kpad) {
  const size_t t = (size_t)blockIdx.x * blockDim.x + threadIdx.x;
  if (t >= (size_t)R * Kpad) return;
  const int col = (int)(t % Kpad);
  const size_t row = t / Kpad;
  float v = 0.0f;
  if (col < Ca) v = A[row * Ca + col];
  else if (col < Ca + Cb) v = Bm[row * Cb + (col - Ca)];
  X[t] = (_Float16)v;
}

// pad f32 [R,K] rows into f16 [R,Kpad]
__global__ void k_pack_rows(const float* __restrict__ X, _Float16* __restrict__ Y,
                            size_t R, int K, int Kpad) {
  const size_t t = (size_t)blockIdx.x * blockDim.x + threadIdx.x;
  if (t >= R * Kpad) return;
  const int c = (int)(t % Kpad);
  const size_t r = t / Kpad;
  Y[t] = (_Float16)(c < K ? X[r * K + c] : 0.0f);
}

// gather sampled coords: out[b,m,:] = xyz[b, idx[b,m], :], optional second dest
__global__ void k_gather_xyz(const float* __restrict__ xyz, const int* __restrict__ idx,
                             int B, int N, int M,
                             float* __restrict__ out, float* __restrict__ out2) {
  const int t = blockIdx.x * blockDim.x + threadIdx.x;
  if (t >= B * M * 3) return;
  const int c = t % 3;
  const int bm = t / 3;
  const int m = bm % M;
  const int b = bm / M;
  const float v = xyz[((size_t)b * N + idx[(size_t)b * M + m]) * 3 + c];
  out[t] = v;
  if (out2) out2[t] = v;
}

__global__ void k_i2f(const int* __restrict__ a, float* __restrict__ o, int n) {
  const int t = blockIdx.x * blockDim.x + threadIdx.x;
  if (t < n) o[t] = (float)a[t];
}

// ---------------------------------------------------------------------------
extern "C" void kernel_launch(void* const* d_in, const int* in_sizes, int n_in,
                              void* d_out, int out_size, void* d_ws, size_t ws_size,
                              hipStream_t stream) {
  (void)in_sizes; (void)n_in; (void)out_size; (void)ws_size;
  const float* pc    = (const float*)d_in[0];
  const float* feat  = (const float*)d_in[1];
  const float* d0_W  = (const float*)d_in[2];   const float* d0_b  = (const float*)d_in[3];
  const float* d1_W0 = (const float*)d_in[4];   const float* d1_b0 = (const float*)d_in[5];
  const float* d1_W1 = (const float*)d_in[6];   const float* d1_b1 = (const float*)d_in[7];
  const float* d2_W0 = (const float*)d_in[8];   const float* d2_b0 = (const float*)d_in[9];
  const float* d2_W1 = (const float*)d_in[10];  const float* d2_b1 = (const float*)d_in[11];
  const float* d3_W0 = (const float*)d_in[12];  const float* d3_b0 = (const float*)d_in[13];
  const float* d3_W1 = (const float*)d_in[14];  const float* d3_b1 = (const float*)d_in[15];
  const float* d4_W0 = (const float*)d_in[16];  const float* d4_b0 = (const float*)d_in[17];
  const float* d4_W1 = (const float*)d_in[18];  const float* d4_b1 = (const float*)d_in[19];
  const float* u4_W1 = (const float*)d_in[20];  const float* u4_b1 = (const float*)d_in[21];
  const float* u4_W2 = (const float*)d_in[22];  const float* u4_b2 = (const float*)d_in[23];
  const float* u3_W1 = (const float*)d_in[24];  const float* u3_b1 = (const float*)d_in[25];
  const float* u3_W2 = (const float*)d_in[26];  const float* u3_b2 = (const float*)d_in[27];
  const float* u2_W1 = (const float*)d_in[28];  const float* u2_b1 = (const float*)d_in[29];
  const float* u2_W2 = (const float*)d_in[30];  const float* u2_b2 = (const float*)d_in[31];
  const float* u1_W1 = (const float*)d_in[32];  const float* u1_b1 = (const float*)d_in[33];
  const float* u1_W2 = (const float*)d_in[34];  const float* u1_b2 = (const float*)d_in[35];

  const int B = 2, N0 = 8192;
  const int M1 = 2048, M2 = 512, M3 = 128, M4 = 64;

  // d_out tuple layout (floats): x1,x2,x3,i1,i2,i3,u0,u1,u2,u3
  float* out = (float*)d_out;
  const size_t o_x1 = 0;
  const size_t o_x2 = o_x1 + (size_t)B * M1 * 3;
  const size_t o_x3 = o_x2 + (size_t)B * M2 * 3;
  const size_t o_i1 = o_x3 + (size_t)B * M3 * 3;
  const size_t o_i2 = o_i1 + (size_t)B * M1;
  const size_t o_i3 = o_i2 + (size_t)B * M2;
  const size_t o_u0 = o_i3 + (size_t)B * M3;
  const size_t o_u1 = o_u0 + (size_t)B * N0 * 32;
  const size_t o_u2 = o_u1 + (size_t)B * M1 * 64;
  const size_t o_u3 = o_u2 + (size_t)B * M2 * 128;

  // ---- workspace bump allocator ----
  char* wsp = (char*)d_ws;
  size_t cur = 0;
  auto alloc = [&](size_t bytes) -> void* {
    cur = (cur + 255) & ~(size_t)255;
    void* p = wsp + cur;
    cur += bytes;
    return p;
  };

  float* f0 = (float*)alloc((size_t)B * N0 * 32 * 4);
  float* f1 = (float*)alloc((size_t)B * M1 * 64 * 4);
  float* f2 = (float*)alloc((size_t)B * M2 * 128 * 4);
  float* f3 = (float*)alloc((size_t)B * M3 * 192 * 4);
  float* f4 = (float*)alloc((size_t)B * M4 * 192 * 4);
  float* x1 = (float*)alloc((size_t)B * M1 * 3 * 4);
  float* x2 = (float*)alloc((size_t)B * M2 * 3 * 4);
  float* x3 = (float*)alloc((size_t)B * M3 * 3 * 4);
  float* x4 = (float*)alloc((size_t)B * M4 * 3 * 4);
  float* pool = (float*)alloc((size_t)B * N0 * 32 * 4);
  int* i1w = (int*)alloc((size_t)B * M1 * 4);
  int* i2w = (int*)alloc((size_t)B * M2 * 4);
  int* i3w = (int*)alloc((size_t)B * M3 * 4);
  int* i4w = (int*)alloc((size_t)B * M4 * 4);
  int* nidx = (int*)alloc((size_t)B * N0 * 8 * 4);
  _Float16* bufA = (_Float16*)alloc((size_t)131072 * 96 * 2);  // grouped X / concat X2
  _Float16* bufB = (_Float16*)alloc((size_t)131072 * 32 * 2);  // h1
  _Float16* bufC = (_Float16*)alloc((size_t)65536 * 64 * 2);   // h2

  auto packW = [&](const float* W, int Cin, int Cout, int Kpad) -> _Float16* {
    _Float16* wh = (_Float16*)alloc((size_t)Kpad * Cout * sizeof(_Float16));
    const size_t n = (size_t)Kpad * Cout;
    k_pack_w_frag<<<cdivh(n, 256), 256, 0, stream>>>(W, wh, Cin, Cout, Kpad / 32);
    return wh;
  };
  _Float16* w_d0   = packW(d0_W, 3, 32, 32);
  _Float16* w_d1_0 = packW(d1_W0, 35, 32, 64);
  _Float16* w_d1_1 = packW(d1_W1, 32, 64, 32);
  _Float16* w_d2_0 = packW(d2_W0, 67, 64, 96);
  _Float16* w_d2_1 = packW(d2_W1, 64, 128, 64);
  _Float16* w_d3_0 = packW(d3_W0, 131, 128, 160);
  _Float16* w_d3_1 = packW(d3_W1, 128, 192, 128);
  _Float16* w_d4_0 = packW(d4_W0, 195, 192, 224);
  _Float16* w_d4_1 = packW(d4_W1, 192, 192, 192);
  _Float16* w_u4_1 = packW(u4_W1, 195, 192, 224);
  _Float16* w_u4_2 = packW(u4_W2, 384, 192, 384);
  _Float16* w_u3_1 = packW(u3_W1, 195, 128, 224);
  _Float16* w_u3_2 = packW(u3_W2, 256, 128, 256);
  _Float16* w_u2_1 = packW(u2_W1, 131, 64, 160);
  _Float16* w_u2_2 = packW(u2_W2, 128, 64, 128);
  _Float16* w_u1_1 = packW(u1_W1, 67, 32, 96);
  _Float16* w_u1_2 = packW(u1_W2, 64, 32, 64);

  // K-templated GEMM dispatch (write f16 XOR f32)
  auto gemm = [&](const _Float16* X, const _Float16* Wp, const float* bias,
                  _Float16* Yh, float* Yf, int R, int K, int N) {
    dim3 grid(R / 16, N / 32, 1);
    switch (K / 32) {
#define GEMM_CASE(KT)                                                                    \
      case KT:                                                                           \
        if (Yh) k_wmma_gemm2<KT, true, false><<<grid, 32, 0, stream>>>(X, Wp, bias, Yh, nullptr, R, N); \
        else    k_wmma_gemm2<KT, false, true><<<grid, 32, 0, stream>>>(X, Wp, bias, nullptr, Yf, R, N); \
        break;
      GEMM_CASE(1) GEMM_CASE(2) GEMM_CASE(3) GEMM_CASE(4) GEMM_CASE(5)
      GEMM_CASE(6) GEMM_CASE(7) GEMM_CASE(8) GEMM_CASE(12)
#undef GEMM_CASE
      default: break;
    }
  };

  // ---- d0: f0 = relu(feat @ d0_W + b) ----
  {
    const size_t R = (size_t)B * N0;
    k_pack_rows<<<cdivh(R * 32, 256), 256, 0, stream>>>(feat, bufA, R, 3, 32);
    gemm(bufA, w_d0, d0_b, nullptr, f0, (int)R, 32, 32);
  }

  // ---- SetConv down level ----
  auto setconv = [&](const float* xyz_in, int Nin, const float* feat_in, int Cf,
                     int M, int* fidx, float* nxyz, float* nxyz_out2,
                     _Float16* w0, const float* b0, int C1,
                     _Float16* w1, const float* b1, int C2,
                     float* fout, int Kpad) {
    const size_t sh = (size_t)Nin * 4 + 256 * (4 + 4);
    k_fps<<<B, 256, sh, stream>>>(xyz_in, Nin, M, fidx);
    k_gather_xyz<<<cdivh((size_t)B * M * 3, 256), 256, 0, stream>>>(
        xyz_in, fidx, B, Nin, M, nxyz, nxyz_out2);
    k_knn<16><<<cdivh((size_t)B * M, 128), 128, 0, stream>>>(nxyz, xyz_in, B, M, Nin, nidx);
    const size_t rows = (size_t)B * M * 16;
    k_group<<<cdivh(rows * Kpad, 256), 256, 0, stream>>>(
        xyz_in, feat_in, Cf, nxyz, nidx, B, M, 16, Nin, bufA, Kpad);
    gemm(bufA, w0, b0, bufB, nullptr, (int)rows, Kpad, C1);
    gemm(bufB, w1, b1, bufC, nullptr, (int)rows, C1, C2);
    k_maxpool<<<cdivh((size_t)B * M * C2, 256), 256, 0, stream>>>(bufC, fout, B * M, 16, C2);
  };

  setconv(pc, N0, f0, 32, M1, i1w, x1, out + o_x1, w_d1_0, d1_b0, 32, w_d1_1, d1_b1, 64,  f1, 64);
  setconv(x1, M1, f1, 64, M2, i2w, x2, out + o_x2, w_d2_0, d2_b0, 64, w_d2_1, d2_b1, 128, f2, 96);
  setconv(x2, M2, f2, 128, M3, i3w, x3, out + o_x3, w_d3_0, d3_b0, 128, w_d3_1, d3_b1, 192, f3, 160);
  setconv(x3, M3, f3, 192, M4, i4w, x4, nullptr,    w_d4_0, d4_b0, 192, w_d4_1, d4_b1, 192, f4, 224);

  k_i2f<<<cdivh((size_t)B * M1, 256), 256, 0, stream>>>(i1w, out + o_i1, B * M1);
  k_i2f<<<cdivh((size_t)B * M2, 256), 256, 0, stream>>>(i2w, out + o_i2, B * M2);
  k_i2f<<<cdivh((size_t)B * M3, 256), 256, 0, stream>>>(i3w, out + o_i3, B * M3);

  // ---- SetUpConv level ----
  auto upconv = [&](const float* cxyz, int Nc, const float* cfeat, int Cc,
                    const float* fxyz, int Mf, const float* ffeat, int Cskip,
                    _Float16* w1, const float* b1, int C1,
                    _Float16* w2, const float* b2, int C2,
                    float* outp, int Kpad1, int Kpad2) {
    k_knn<8><<<cdivh((size_t)B * Mf, 128), 128, 0, stream>>>(fxyz, cxyz, B, Mf, Nc, nidx);
    const size_t rows = (size_t)B * Mf * 8;
    k_group<<<cdivh(rows * Kpad1, 256), 256, 0, stream>>>(
        cxyz, cfeat, Cc, fxyz, nidx, B, Mf, 8, Nc, bufA, Kpad1);
    gemm(bufA, w1, b1, bufB, nullptr, (int)rows, Kpad1, C1);
    k_maxpool<<<cdivh((size_t)B * Mf * C1, 256), 256, 0, stream>>>(bufB, pool, B * Mf, 8, C1);
    k_concat2<<<cdivh((size_t)B * Mf * Kpad2, 256), 256, 0, stream>>>(
        pool, C1, ffeat, Cskip, bufA, B * Mf, Kpad2);
    gemm(bufA, w2, b2, nullptr, outp, B * Mf, Kpad2, C2);
  };

  upconv(x4, M4, f4, 192, x3, M3, f3, 192, w_u4_1, u4_b1, 192, w_u4_2, u4_b2, 192,
         out + o_u3, 224, 384);
  upconv(x3, M3, out + o_u3, 192, x2, M2, f2, 128, w_u3_1, u3_b1, 128, w_u3_2, u3_b2, 128,
         out + o_u2, 224, 256);
  upconv(x2, M2, out + o_u2, 128, x1, M1, f1, 64, w_u2_1, u2_b1, 64, w_u2_2, u2_b2, 64,
         out + o_u1, 160, 128);
  upconv(x1, M1, out + o_u1, 64, pc, N0, f0, 32, w_u1_1, u1_b1, 32, w_u1_2, u1_b2, 32,
         out + o_u0, 96, 64);
}
